// VQ_VAE_46188078301439
// MI455X (gfx1250) — compile-verified, matrix operand order checked by reading, not run on hardware
//
#include <hip/hip_runtime.h>
#include <hip/hip_bf16.h>

// ---------------------------------------------------------------------------
// VQ-VAE forward for MI455X (gfx1250). Every conv / convT / codebook matmul is
// one implicit-GEMM kernel on v_wmma_f32_16x16x32_bf16 (bf16 in, f32 acc).
// 128x64 output tile, 8 waves, 4 WMMA per wave per 32-K step. Weight tiles are
// staged LDS-direct with GLOBAL_LOAD_ASYNC_TO_LDS_B128 (ASYNCcnt) when the
// toolchain exposes the builtin; activations are gathered (implicit im2col,
// with lhs_dilation so convT == conv with flipped packed weights).
// ---------------------------------------------------------------------------

typedef __attribute__((ext_vector_type(16))) __bf16 v16bf;
typedef __attribute__((ext_vector_type(8)))  __bf16 v8bf;
typedef __attribute__((ext_vector_type(8)))  float  v8f;
typedef __attribute__((ext_vector_type(4)))  int    v4i;

#define TM 128
#define TN 64
#define TK 32

#if defined(__gfx1250__) && \
    __has_builtin(__builtin_amdgcn_global_load_async_to_lds_b128) && \
    __has_builtin(__builtin_amdgcn_s_wait_asynccnt)
#define USE_ASYNC_LDS 1
#else
#define USE_ASYNC_LDS 0
#endif

// flags: 1 = relu on A input, 2 = relu on output, 4 = residual add (res ptr,
// same MxN layout, may alias out), 8 = per-N bias add.
__global__ __launch_bounds__(256) void k_gemm_conv_wmma(
    const float* __restrict__ src, const __bf16* __restrict__ wpk,
    const float* __restrict__ bias, const float* __restrict__ res,
    float* __restrict__ out,
    int M, int N, int K,
    int Hin, int Win, int Cin, int Hout, int Wout,
    int KH, int KW, int S, int P, int L, int flags)
{
  __shared__ __bf16 As[TM][TK];   // 8 KiB
  __shared__ __bf16 Bs[TN][TK];   // 4 KiB

  const int t    = threadIdx.x;
  const int lane = t & 31;
  const int wave = t >> 5;        // 0..7 : one 16-row strip each

  const int mtile = blockIdx.x * TM;
  const int ntile = blockIdx.y * TN;

  // cooperative-load mapping: thread t owns rows (t>>2) and (t>>2)+64 of A,
  // row (t>>2) of B, 8 consecutive k (one 16-byte granule) each.
  const int arow  = t >> 2;           // 0..63
  const int akoff = (t & 3) * 8;      // 0,8,16,24

  int mb[2], myy[2], mxx[2]; bool mv[2];
  #pragma unroll
  for (int rr = 0; rr < 2; ++rr) {
    int m = mtile + arow + rr * 64;
    mv[rr] = (m < M);
    int q = mv[rr] ? m : 0;
    mxx[rr] = q % Wout; q /= Wout; myy[rr] = q % Hout; mb[rr] = q / Hout;
  }
  const int bn  = ntile + arow;
  const int bnc = (bn < N) ? bn : 0;          // clamp: garbage rows never stored

  const int khkw = KH * KW;
  const int dilH = (Hin - 1) * L + 1;   // lhs-dilated input extent
  const int dilW = (Win - 1) * L + 1;

  v8f acc[4] = {v8f{}, v8f{}, v8f{}, v8f{}};

  for (int k0 = 0; k0 < K; k0 += TK) {
    // k-granule clamp (K is always a multiple of 8 here); invalid granules
    // load valid-but-wrong weights which multiply zeroed A columns.
    const int kg = k0 + akoff;
    const int kc = (kg + 8 <= K) ? kg : 0;
    const __bf16* wsrc = wpk + (long)bnc * K + kc;

#if USE_ASYNC_LDS
    // ---- B tile: async DMA straight into LDS (tracked by ASYNCcnt) ----
    __builtin_amdgcn_global_load_async_to_lds_b128(
        (__attribute__((address_space(1))) v4i*)wsrc,
        (__attribute__((address_space(3))) v4i*)&Bs[arow][akoff], 0, 0);
#else
    *(v8bf*)&Bs[arow][akoff] = *(const v8bf*)wsrc;
#endif
    if (k0 + TK < K)  // pull next weight tile toward L2 (global_prefetch_b8)
      __builtin_prefetch((const void*)(wsrc + TK), 0, 0);

    // ---- A tile: implicit im2col gather (f32 -> bf16), relu_in optional ----
    #pragma unroll
    for (int rr = 0; rr < 2; ++rr) {
      #pragma unroll
      for (int i = 0; i < 8; ++i) {
        int k = kg + i;
        float v = 0.f;
        if (mv[rr] && k < K) {
          int ci = k / khkw; int r = k - ci * khkw;
          int ky = r / KW;   int kx = r - ky * KW;
          int ty = myy[rr] * S - P + ky;
          int tx = mxx[rr] * S - P + kx;
          if (ty >= 0 && ty < dilH && tx >= 0 && tx < dilW) {
            int iy = ty, ix = tx;
            bool ok = true;
            if (L != 1) { ok = ((ty % L) == 0) && ((tx % L) == 0); iy = ty / L; ix = tx / L; }
            if (ok) v = src[(((long)mb[rr] * Hin + iy) * Win + ix) * Cin + ci];
          }
          if (flags & 1) v = fmaxf(v, 0.f);
        }
        As[arow + rr * 64][akoff + i] = (__bf16)v;
      }
    }

#if USE_ASYNC_LDS
    __builtin_amdgcn_s_wait_asynccnt(0);
#endif
    __syncthreads();

    // ---- fragments per the ISA 16-bit striping:
    // lane<16: row=lane,    halves K{0..7} then K{16..23}
    // lane>=16: row=lane-16, halves K{8..15} then K{24..31}
    const int lrow = lane & 15;
    const int h8   = (lane >> 4) * 8;
    const __bf16* ap = &As[wave * 16 + lrow][h8];
    v8bf alo = *(const v8bf*)ap;
    v8bf ahi = *(const v8bf*)(ap + 16);
    v16bf af = __builtin_shufflevector(alo, ahi, 0,1,2,3,4,5,6,7,8,9,10,11,12,13,14,15);
    #pragma unroll
    for (int j = 0; j < 4; ++j) {
      const __bf16* bp = &Bs[j * 16 + lrow][h8];
      v8bf bl = *(const v8bf*)bp;
      v8bf bh = *(const v8bf*)(bp + 16);
      v16bf bfj = __builtin_shufflevector(bl, bh, 0,1,2,3,4,5,6,7,8,9,10,11,12,13,14,15);
      acc[j] = __builtin_amdgcn_wmma_f32_16x16x32_bf16(false, af, false, bfj,
                                                       (short)0, acc[j], false, false);
    }
    __syncthreads();
  }

  // ---- epilogue: C/D layout lane = n, vgpr r = row (lane>=16 -> +8) ----
  const int el = lane & 15;
  const int mo = (lane >> 4) * 8;
  #pragma unroll
  for (int j = 0; j < 4; ++j) {
    int n = ntile + j * 16 + el;
    if (n >= N) continue;
    float bv = (flags & 8) ? bias[n] : 0.f;
    #pragma unroll
    for (int r = 0; r < 8; ++r) {
      int mm = mtile + wave * 16 + mo + r;
      if (mm >= M) continue;
      float v = acc[j][r] + bv;
      if (flags & 4) v += res[(long)mm * N + n];
      if (flags & 2) v = fmaxf(v, 0.f);
      out[(long)mm * N + n] = v;
    }
  }
}

// ---------------------------------------------------------------------------
// Support kernels
// ---------------------------------------------------------------------------

__global__ void k_nchw_to_nhwc(const float* __restrict__ in, float* __restrict__ out,
                               int B, int C, int H, int W) {
  long i = (long)blockIdx.x * blockDim.x + threadIdx.x;
  long total = (long)B * C * H * W;
  if (i >= total) return;
  int w = (int)(i % W); long q = i / W;
  int h = (int)(q % H); q /= H;
  int c = (int)(q % C); int b = (int)(q / C);
  out[(((long)b * H + h) * W + w) * C + c] = in[i];
}

__global__ void k_nhwc_to_nchw(const float* __restrict__ in, float* __restrict__ out,
                               int B, int C, int H, int W) {
  long i = (long)blockIdx.x * blockDim.x + threadIdx.x;   // NCHW index
  long total = (long)B * C * H * W;
  if (i >= total) return;
  int w = (int)(i % W); long q = i / W;
  int h = (int)(q % H); q /= H;
  int c = (int)(q % C); int b = (int)(q / C);
  out[i] = in[(((long)b * H + h) * W + w) * C + c];
}

// mode 0: w is (Cout, Cin, KH, KW) OIHW  ->  B[n][k]= w[n][ci][ky][kx]
// mode 1: w is (Cin, Cout, KH, KW) convT ->  B[n][k]= w[ci][n][KH-1-ky][KW-1-kx]
__global__ void k_pack_w(const float* __restrict__ w, __bf16* __restrict__ outw,
                         int Cout, int Cin, int KH, int KW, int mode, float scale) {
  int i = blockIdx.x * blockDim.x + threadIdx.x;
  int K = Cin * KH * KW;
  int total = Cout * K;
  if (i >= total) return;
  int n = i / K; int k = i - n * K;
  int ci = k / (KH * KW); int r = k % (KH * KW);
  int ky = r / KW;        int kx = r % KW;
  float v;
  if (mode == 0) v = w[(((long)n * Cin + ci) * KH + ky) * KW + kx];
  else           v = w[(((long)ci * Cout + n) * KH + (KH - 1 - ky)) * KW + (KW - 1 - kx)];
  outw[i] = (__bf16)(v * scale);
}

__global__ void k_emb_halfnorm(const float* __restrict__ emb, float* __restrict__ embn,
                               int Ne, int D) {
  int n = blockIdx.x * blockDim.x + threadIdx.x;
  if (n >= Ne) return;
  float s = 0.f;
  for (int d = 0; d < D; ++d) { float e = emb[(long)n * D + d]; s += e * e; }
  embn[n] = 0.5f * s;
}

__global__ void k_zero_f32(float* p, long n) {
  long i = (long)blockIdx.x * blockDim.x + threadIdx.x;
  if (i < n) p[i] = 0.f;
}

// one wave32 per row: argmin over N score columns
__global__ __launch_bounds__(256) void k_argmin_rows(const float* __restrict__ sc,
                                                     int* __restrict__ idx, int M, int N) {
  int row = blockIdx.x * 8 + (threadIdx.x >> 5);
  int lane = threadIdx.x & 31;
  if (row >= M) return;
  const float* r = sc + (long)row * N;
  float best = 3.4e38f; int bi = 0;
  for (int c = lane; c < N; c += 32) {
    float v = r[c];
    if (v < best) { best = v; bi = c; }
  }
  for (int off = 16; off; off >>= 1) {
    float ob = __shfl_xor(best, off, 32);
    int   oi = __shfl_xor(bi,   off, 32);
    if (ob < best || (ob == best && oi < bi)) { best = ob; bi = oi; }
  }
  if (lane == 0) idx[row] = bi;
}

__global__ void k_gather_loss(const float* __restrict__ z, const float* __restrict__ emb,
                              const int* __restrict__ idx, float* __restrict__ zq,
                              float* __restrict__ counts, float* __restrict__ lossacc,
                              int M, int D) {
  int m = blockIdx.x * blockDim.x + threadIdx.x;
  if (m >= M) return;
  int e = idx[m];
  const float* ev = emb + (long)e * D;
  const float* zv = z   + (long)m * D;
  float*       qv = zq  + (long)m * D;
  float s = 0.f;
  for (int d = 0; d < D; ++d) {
    float q = ev[d];
    float diff = q - zv[d];
    s += diff * diff;
    qv[d] = q;               // straight-through forward value == zq
  }
  atomicAdd(lossacc, s);
  atomicAdd(&counts[e], 1.0f);
}

__global__ void k_finalize(const float* __restrict__ counts, const float* __restrict__ lossacc,
                           float* __restrict__ out2, float invMD, float invM) {
  __shared__ float red[512];
  int t = threadIdx.x;
  float em = counts[t] * invM;
  red[t] = em * logf(em + 1e-10f);
  __syncthreads();
  for (int s = 256; s; s >>= 1) { if (t < s) red[t] += red[t + s]; __syncthreads(); }
  if (t == 0) {
    out2[0] = lossacc[0] * invMD * 1.25f;   // (1 + BETA) * mse
    out2[1] = expf(-red[0]);
  }
}

// ---------------------------------------------------------------------------
// Host orchestration
// ---------------------------------------------------------------------------

static inline void gemm(hipStream_t s, const float* src, const __bf16* w,
                        const float* bias, const float* res, float* out,
                        int M, int N, int K,
                        int Hin, int Win, int Cin, int Hout, int Wout,
                        int KH, int KW, int S, int P, int L, int flags) {
  dim3 g((M + TM - 1) / TM, (N + TN - 1) / TN);
  k_gemm_conv_wmma<<<g, dim3(256), 0, s>>>(src, w, bias, res, out, M, N, K,
                                           Hin, Win, Cin, Hout, Wout,
                                           KH, KW, S, P, L, flags);
}

static inline void pack(hipStream_t s, const float* w, __bf16* dst,
                        int Cout, int Cin, int KH, int KW, int mode, float scale) {
  int total = Cout * Cin * KH * KW;
  k_pack_w<<<(total + 255) / 256, 256, 0, s>>>(w, dst, Cout, Cin, KH, KW, mode, scale);
}

extern "C" void kernel_launch(void* const* d_in, const int* in_sizes, int n_in,
                              void* d_out, int out_size, void* d_ws, size_t ws_size,
                              hipStream_t stream) {
  // ---- inputs (setup_inputs dict order: x, then params in insertion order) ----
  const float* x        = (const float*)d_in[0];
  const float* enc_w1   = (const float*)d_in[1];
  const float* enc_b1   = (const float*)d_in[2];
  const float* enc_w2   = (const float*)d_in[3];
  const float* enc_b2   = (const float*)d_in[4];
  const float* enc_w3   = (const float*)d_in[5];
  const float* enc_b3   = (const float*)d_in[6];
  const float* enc_r0w1 = (const float*)d_in[7];
  const float* enc_r0w2 = (const float*)d_in[8];
  const float* enc_r1w1 = (const float*)d_in[9];
  const float* enc_r1w2 = (const float*)d_in[10];
  const float* down_w   = (const float*)d_in[11];
  const float* down_b   = (const float*)d_in[12];
  const float* emb      = (const float*)d_in[13];
  const float* up_w     = (const float*)d_in[14];
  const float* up_b     = (const float*)d_in[15];
  const float* dec_r0w1 = (const float*)d_in[16];
  const float* dec_r0w2 = (const float*)d_in[17];
  const float* dec_r1w1 = (const float*)d_in[18];
  const float* dec_r1w2 = (const float*)d_in[19];
  const float* dec_w1   = (const float*)d_in[20];
  const float* dec_b1   = (const float*)d_in[21];
  const float* dec_w2   = (const float*)d_in[22];
  const float* dec_b2   = (const float*)d_in[23];

  // ---- geometry ----
  const int B = 16;
  const int M1 = 16 * 128 * 128;    // 262144
  const int M2 = 16 * 64 * 64;      // 65536
  const int M3 = 16 * 256 * 256;    // 1048576
  const int D  = 128;               // EMB_DIM
  const int NE = 512;               // N_EMB

  // ---- workspace layout (bytes) ----
  char* ws = (char*)d_ws;
  float*  bufA   = (float*)(ws + 0);                         // 262144*64 f32 = 64 MiB
  float*  bufB   = (float*)(ws + (size_t)67108864);          // 65536*128 f32 = 32 MiB
  float*  bufX   = (float*)(ws + (size_t)100663296);         // 1048576*3 f32 = 12 MiB
  float*  scores = (float*)(ws + (size_t)113246208);         // 65536*512 f32 = 128 MiB
  int*    idx    = (int*)  (ws + (size_t)247463936);         // 65536 i32
  float*  counts = (float*)(ws + (size_t)247726080);         // 512 f32
  float*  embn   = (float*)(ws + (size_t)247728128);         // 512 f32
  float*  lossa  = (float*)(ws + (size_t)247730176);         // 1 f32
  __bf16* wp     = (__bf16*)(ws + (size_t)247730432);        // packed weights (~1.6 MiB)

  // packed-weight offsets (halves)
  __bf16* W1   = wp;                 // 64 x 48
  __bf16* W2   = W1   + 3072;        // 128 x 1024
  __bf16* W3   = W2   + 131072;      // 128 x 1152
  __bf16* R0A  = W3   + 147456;      // 32 x 1152
  __bf16* R0B  = R0A  + 36864;       // 128 x 32
  __bf16* R1A  = R0B  + 4096;        // 32 x 1152
  __bf16* R1B  = R1A  + 36864;       // 128 x 32
  __bf16* DW   = R1B  + 4096;        // 128 x 128
  __bf16* EMBW = DW   + 16384;       // 512 x 128 (scaled by -1)
  __bf16* UPW  = EMBW + 65536;       // 128 x 1152 (convT flip)
  __bf16* DR0A = UPW  + 147456;      // 32 x 1152
  __bf16* DR0B = DR0A + 36864;       // 128 x 32
  __bf16* DR1A = DR0B + 4096;        // 32 x 1152
  __bf16* DR1B = DR1A + 36864;       // 128 x 32
  __bf16* DW1  = DR1B + 4096;        // 64 x 2048 (convT flip)
  __bf16* DW2  = DW1  + 131072;      // 3 x 1024  (convT flip)

  // ---- weight packing + codebook prep (runs every call; deterministic) ----
  pack(stream, enc_w1,   W1,   64, 3,   4, 4, 0,  1.f);
  pack(stream, enc_w2,   W2,   128, 64, 4, 4, 0,  1.f);
  pack(stream, enc_w3,   W3,   128, 128,3, 3, 0,  1.f);
  pack(stream, enc_r0w1, R0A,  32, 128, 3, 3, 0,  1.f);
  pack(stream, enc_r0w2, R0B,  128, 32, 1, 1, 0,  1.f);
  pack(stream, enc_r1w1, R1A,  32, 128, 3, 3, 0,  1.f);
  pack(stream, enc_r1w2, R1B,  128, 32, 1, 1, 0,  1.f);
  pack(stream, down_w,   DW,   128, 128,1, 1, 0,  1.f);
  pack(stream, emb,      EMBW, 512, 128,1, 1, 0, -1.f);   // so acc = -z.e
  pack(stream, up_w,     UPW,  128, 128,3, 3, 1,  1.f);   // convT flip+transpose
  pack(stream, dec_r0w1, DR0A, 32, 128, 3, 3, 0,  1.f);
  pack(stream, dec_r0w2, DR0B, 128, 32, 1, 1, 0,  1.f);
  pack(stream, dec_r1w1, DR1A, 32, 128, 3, 3, 0,  1.f);
  pack(stream, dec_r1w2, DR1B, 128, 32, 1, 1, 0,  1.f);
  pack(stream, dec_w1,   DW1,  64, 128, 4, 4, 1,  1.f);
  pack(stream, dec_w2,   DW2,  3,  64,  4, 4, 1,  1.f);
  k_emb_halfnorm<<<2, 256, 0, stream>>>(emb, embn, NE, D);
  k_zero_f32<<<2, 256, 0, stream>>>(counts, 512);
  k_zero_f32<<<1, 1,   0, stream>>>(lossa, 1);

  // ---- x: NCHW -> NHWC ----
  k_nchw_to_nhwc<<<(M3 * 3 + 255) / 256, 256, 0, stream>>>(x, bufX, B, 3, 256, 256);

  // ---- encoder ----
  // enc1: 4x4 s2 p1, 3->64, relu_out+bias
  gemm(stream, bufX, W1, enc_b1, nullptr, bufA, M1, 64, 48,
       256, 256, 3, 128, 128, 4, 4, 2, 1, 1, /*flags*/ 2 | 8);
  // enc2: 4x4 s2 p1, 64->128, relu_out+bias
  gemm(stream, bufA, W2, enc_b2, nullptr, bufB, M2, 128, 1024,
       128, 128, 64, 64, 64, 4, 4, 2, 1, 1, 2 | 8);
  // enc3: 3x3 s1 p1, 128->128, bias only
  gemm(stream, bufB, W3, enc_b3, nullptr, bufA, M2, 128, 1152,
       64, 64, 128, 64, 64, 3, 3, 1, 1, 1, 8);
  // res0: t = conv3x3(relu(h)); h += conv1x1(relu(t))
  gemm(stream, bufA, R0A, nullptr, nullptr, bufB, M2, 32, 1152,
       64, 64, 128, 64, 64, 3, 3, 1, 1, 1, 1);
  gemm(stream, bufB, R0B, nullptr, bufA, bufA, M2, 128, 32,
       64, 64, 32, 64, 64, 1, 1, 1, 0, 1, 1 | 4);
  // res1
  gemm(stream, bufA, R1A, nullptr, nullptr, bufB, M2, 32, 1152,
       64, 64, 128, 64, 64, 3, 3, 1, 1, 1, 1);
  gemm(stream, bufB, R1B, nullptr, bufA, bufA, M2, 128, 32,
       64, 64, 32, 64, 64, 1, 1, 1, 0, 1, 1 | 4);
  // down: z = conv1x1(relu(h)) + b   -> bufB
  gemm(stream, bufA, DW, down_b, nullptr, bufB, M2, 128, 128,
       64, 64, 128, 64, 64, 1, 1, 1, 0, 1, 1 | 8);

  // ---- quantize: scores = 0.5|e|^2 - z.e  (B packed as -emb, bias = 0.5|e|^2)
  gemm(stream, bufB, EMBW, embn, nullptr, scores, M2, NE, 128,
       64, 64, 128, 64, 64, 1, 1, 1, 0, 1, 8);
  k_argmin_rows<<<(M2 + 7) / 8, 256, 0, stream>>>(scores, idx, M2, NE);
  // zq gather (-> bufA), loss accumulation, histogram
  k_gather_loss<<<(M2 + 255) / 256, 256, 0, stream>>>(bufB, emb, idx, bufA,
                                                      counts, lossa, M2, D);

  // ---- decoder ----
  // up: convT 3x3 s1 p1 == conv(flip(w)) p1; 128->128, bias
  gemm(stream, bufA, UPW, up_b, nullptr, bufB, M2, 128, 1152,
       64, 64, 128, 64, 64, 3, 3, 1, 1, 1, 8);
  // dec res0
  gemm(stream, bufB, DR0A, nullptr, nullptr, bufA, M2, 32, 1152,
       64, 64, 128, 64, 64, 3, 3, 1, 1, 1, 1);
  gemm(stream, bufA, DR0B, nullptr, bufB, bufB, M2, 128, 32,
       64, 64, 32, 64, 64, 1, 1, 1, 0, 1, 1 | 4);
  // dec res1
  gemm(stream, bufB, DR1A, nullptr, nullptr, bufA, M2, 32, 1152,
       64, 64, 128, 64, 64, 3, 3, 1, 1, 1, 1);
  gemm(stream, bufA, DR1B, nullptr, bufB, bufB, M2, 128, 32,
       64, 64, 32, 64, 64, 1, 1, 1, 0, 1, 1 | 4);
  // dec convT1: 4x4 s2 p1 -> conv(flip) pad2, lhs_dilation 2; relu_in+relu_out+bias
  gemm(stream, bufB, DW1, dec_b1, nullptr, bufA, M1, 64, 2048,
       64, 64, 128, 128, 128, 4, 4, 1, 2, 2, 1 | 2 | 8);
  // dec convT2: 4x4 s2 p1 -> conv(flip) pad2, lhs_dilation 2; bias only
  gemm(stream, bufA, DW2, dec_b2, nullptr, bufX, M3, 3, 1024,
       128, 128, 64, 256, 256, 4, 4, 1, 2, 2, 8);

  // ---- outputs ----
  float* out = (float*)d_out;
  k_nhwc_to_nchw<<<(M3 * 3 + 255) / 256, 256, 0, stream>>>(bufX, out, B, 3, 256, 256);
  k_finalize<<<1, 512, 0, stream>>>(counts, lossa, out + (long)M3 * 3,
                                    1.0f / ((float)M2 * (float)D), 1.0f / (float)M2);
}
